// AttentionMixer_39376260170426
// MI455X (gfx1250) — matrix operand order repaired and checked by reading, not compiled, not run on hardware
//
#include <hip/hip_runtime.h>
#include <hip/hip_bf16.h>

// Problem constants (from reference): B=2, H=16, S=2048, D=64
#define SEQ   2048
#define DIM   64
#define NBH   32          // B*H
#define ROWS  16          // query rows per wave
#define WAVES 4           // waves per block

typedef __attribute__((ext_vector_type(16))) __bf16 v16bf;
typedef __attribute__((ext_vector_type(8)))  float  v8f;
typedef __attribute__((ext_vector_type(4)))  float  v4f;

// ---------------------------------------------------------------------------
// Prepass: vt[bh][d][j] = bf16(v[bh][j][d])  (tiled transpose + convert)
// Makes each WMMA B-fragment a contiguous 32B load per lane. vt (8 MB) is
// reused by 128 row-blocks per head, so we want it resident in L2.
// ---------------------------------------------------------------------------
__global__ __launch_bounds__(256)
void vt_transpose_kernel(const float* __restrict__ v, __bf16* __restrict__ vt) {
    __shared__ float tile[16][17];
    const int bh = blockIdx.z;
    const int j0 = blockIdx.x * 16;
    const int d0 = blockIdx.y * 16;
    const int tx = threadIdx.x, ty = threadIdx.y;
    tile[ty][tx] = v[((size_t)bh * SEQ + (j0 + ty)) * DIM + (d0 + tx)];
    __syncthreads();
    vt[((size_t)bh * DIM + (d0 + ty)) * SEQ + (j0 + tx)] = (__bf16)tile[tx][ty];
}

// ---------------------------------------------------------------------------
// Main: online-softmax (flash style) + bf16 WMMA accumulation.
// One wave owns a 16-row query block; K processed in chunks of 32.
// ---------------------------------------------------------------------------
__global__ __launch_bounds__(WAVES * 32)
void attn_mixer_kernel(const float* __restrict__ logits,
                       const __bf16* __restrict__ vt,
                       float* __restrict__ out) {
    __shared__ float s_scale[WAVES][16];
    __shared__ float s_l[WAVES][16];

    const int lane = threadIdx.x & 31;
    const int wid  = threadIdx.x >> 5;
    const int w    = blockIdx.x * WAVES + wid;      // global wave id
    const int bh   = w / (SEQ / ROWS);
    const int rb   = w % (SEQ / ROWS);
    const int row_base = rb * ROWS;

    const int h   = lane >> 4;   // lane half (A: K-half select; C/D: M-half)
    const int r16 = lane & 15;   // A: row M; B/C/D: column N

    const float*  lg  = logits + ((size_t)bh * SEQ + row_base + r16) * SEQ;
    const __bf16* vtb = vt + (size_t)bh * DIM * SEQ;

    v8f acc0 = {}, acc1 = {}, acc2 = {}, acc3 = {};
    float mx = -INFINITY;   // running row max (raw logit domain)
    float l  = 0.0f;        // running row sum of exp
    const float LOG2E = 1.4426950408889634f;

    for (int j0 = 0; j0 < SEQ; j0 += 32) {
        // --- Issue all global loads for this chunk up front -----------------
        // B fragments: vt row (d) contiguous in j => one 32B load each (RT,
        // stays hot in L2 across row-blocks).
        const __bf16* bp = vtb + (size_t)r16 * SEQ + j0 + h * 16;
        v16bf b0 = *(const v16bf*)(bp);
        v16bf b1 = *(const v16bf*)(bp + 16 * SEQ);
        v16bf b2 = *(const v16bf*)(bp + 32 * SEQ);
        v16bf b3 = *(const v16bf*)(bp + 48 * SEQ);

        // Logits: pure stream, zero reuse -> non-temporal so it doesn't
        // evict vt from L2. Lane half h owns K = {16h..16h+15}.
        v4f x0 = __builtin_nontemporal_load((const v4f*)(lg + j0 + h * 16));
        v4f x1 = __builtin_nontemporal_load((const v4f*)(lg + j0 + h * 16 + 4));
        v4f x2 = __builtin_nontemporal_load((const v4f*)(lg + j0 + h * 16 + 8));
        v4f x3 = __builtin_nontemporal_load((const v4f*)(lg + j0 + h * 16 + 12));
        // Prefetch a few chunks ahead (global_prefetch_b8).
        __builtin_prefetch(lg + j0 + 256, 0, 1);

        float x[16];
        #pragma unroll
        for (int e = 0; e < 4; ++e) {
            x[e]      = x0[e];
            x[4 + e]  = x1[e];
            x[8 + e]  = x2[e];
            x[12 + e] = x3[e];
        }

        // Row max: tree over 16 local values, then partner-lane exchange.
        float t8[8], t4[4], t2[2];
        #pragma unroll
        for (int e = 0; e < 8; ++e) t8[e] = fmaxf(x[e], x[e + 8]);
        #pragma unroll
        for (int e = 0; e < 4; ++e) t4[e] = fmaxf(t8[e], t8[e + 4]);
        t2[0] = fmaxf(t4[0], t4[2]); t2[1] = fmaxf(t4[1], t4[3]);
        float vmax = fmaxf(t2[0], t2[1]);
        vmax = fmaxf(vmax, __shfl_xor(vmax, 16, 32));

        const bool  changed = vmax > mx;
        const float mx_new  = fmaxf(mx, vmax);
        const float m2      = mx_new * LOG2E;

        // p = exp2(x*log2e - m2) via fma; chunk row-sum tree.
        float p[16];
        #pragma unroll
        for (int e = 0; e < 16; ++e) p[e] = exp2f(fmaf(x[e], LOG2E, -m2));
        float s8[8], s4[4];
        #pragma unroll
        for (int e = 0; e < 8; ++e) s8[e] = p[e] + p[e + 8];
        #pragma unroll
        for (int e = 0; e < 4; ++e) s4[e] = s8[e] + s8[e + 4];
        float csum = (s4[0] + s4[2]) + (s4[1] + s4[3]);
        csum += __shfl_xor(csum, 16, 32);

        // Rescale only when some row's max actually moved (~5 of 64 chunks).
        if (__any(changed)) {
            const float scale = exp2f((mx - mx_new) * LOG2E); // 1.0 if row unchanged
            l  = l * scale + csum;
            mx = mx_new;
            // Broadcast per-row scale (row layout) into accumulator (C) layout.
            s_scale[wid][r16] = scale;
            __builtin_amdgcn_wave_barrier();
            float sc[8];
            #pragma unroll
            for (int r = 0; r < 8; ++r) sc[r] = s_scale[wid][h * 8 + r];
            #pragma unroll
            for (int r = 0; r < 8; ++r) {
                acc0[r] *= sc[r]; acc1[r] *= sc[r];
                acc2[r] *= sc[r]; acc3[r] *= sc[r];
            }
        } else {
            l += csum;
        }

        // Pack A fragment (element e -> VGPR e/2 slot e%2 -> ISA K mapping).
        v16bf a;
        #pragma unroll
        for (int e = 0; e < 16; ++e) a[e] = (__bf16)p[e];

        acc0 = __builtin_amdgcn_wmma_f32_16x16x32_bf16(false, a, false, b0, (short)0, acc0, false, false);
        acc1 = __builtin_amdgcn_wmma_f32_16x16x32_bf16(false, a, false, b1, (short)0, acc1, false, false);
        acc2 = __builtin_amdgcn_wmma_f32_16x16x32_bf16(false, a, false, b2, (short)0, acc2, false, false);
        acc3 = __builtin_amdgcn_wmma_f32_16x16x32_bf16(false, a, false, b3, (short)0, acc3, false, false);
    }

    // Final normalize: broadcast row sums into C layout, multiply by rcp, store.
    s_l[wid][r16] = l;
    __builtin_amdgcn_wave_barrier();
    #pragma unroll
    for (int r = 0; r < 8; ++r) {
        const float inv = __builtin_amdgcn_rcpf(s_l[wid][h * 8 + r]);
        const size_t row = (size_t)(row_base + h * 8 + r);
        float* op = out + ((size_t)bh * SEQ + row) * DIM + r16;
        __builtin_nontemporal_store(acc0[r] * inv, op);
        __builtin_nontemporal_store(acc1[r] * inv, op + 16);
        __builtin_nontemporal_store(acc2[r] * inv, op + 32);
        __builtin_nontemporal_store(acc3[r] * inv, op + 48);
    }
}

// ---------------------------------------------------------------------------
extern "C" void kernel_launch(void* const* d_in, const int* in_sizes, int n_in,
                              void* d_out, int out_size, void* d_ws, size_t ws_size,
                              hipStream_t stream) {
    const float* v      = (const float*)d_in[0];  // [B,H,S,D] fp32
    const float* logits = (const float*)d_in[1];  // [B,H,S,S] fp32
    float* out          = (float*)d_out;          // [B,H,S,D] fp32
    __bf16* vt          = (__bf16*)d_ws;          // [BH,D,S] bf16 (8 MB)

    // Prepass: transpose + bf16-convert v.
    dim3 tgrid(SEQ / 16, DIM / 16, NBH);
    dim3 tblock(16, 16);
    vt_transpose_kernel<<<tgrid, tblock, 0, stream>>>(v, vt);

    // Main: 1 wave per 16-row block, 4 waves per workgroup.
    const int total_waves = NBH * (SEQ / ROWS);        // 4096
    const int blocks = total_waves / WAVES;            // 1024
    attn_mixer_kernel<<<blocks, WAVES * 32, 0, stream>>>(logits, vt, out);
}
